// QuantizedGRU_5437428596953
// MI455X (gfx1250) — matrix operand order repaired and checked by reading, not compiled, not run on hardware
//
#include <hip/hip_runtime.h>

// ---------------------------------------------------------------------------
// Bidirectional GRU for MI455X (gfx1250), bf16 WMMA with fp32 accumulation.
// r-gate is dead in the reference -> only z/n gates are computed (2/3 work).
// Recurrence weights staged to LDS via the Tensor Data Mover (TDM).
// ---------------------------------------------------------------------------

typedef __attribute__((ext_vector_type(16))) __bf16 v16bf;
typedef __attribute__((ext_vector_type(8)))  __bf16 v8bf;
typedef __attribute__((ext_vector_type(8)))  float  v8f;
typedef __attribute__((ext_vector_type(4)))  unsigned int v4u;
typedef __attribute__((ext_vector_type(8)))  int v8i;
typedef __attribute__((ext_vector_type(4)))  int v4i;

#define B_SZ 64
#define S_SZ 512
#define I_SZ 512
#define H_SZ 512
#define NWG  16   // persistent recurrence workgroups (8 per direction)

// ---- WMMA helper ----------------------------------------------------------
__device__ __forceinline__ v8f wmma_bf16(v16bf a, v16bf b, v8f c) {
  // 8 args: (neg_a, A, neg_b, B, c_mod, C, reuse_a, reuse_b)
  return __builtin_amdgcn_wmma_f32_16x16x32_bf16(false, a, false, b, (short)0, c,
                                                 false, false);
}

// A-matrix fragment (16x32 bf16): lane<16 holds K chunks {0-7,16-23} of the
// 32-wide K window; lane>=16 holds {8-15,24-31} (ISA 7.12.2 layout).
__device__ __forceinline__ v16bf fragA_bf16(const __bf16* __restrict__ row,
                                            int kiter, int half) {
  const int base = kiter * 32 + half * 8;
  v16bf r;
  *reinterpret_cast<v8bf*>(&r)       = *reinterpret_cast<const v8bf*>(row + base);
  *(reinterpret_cast<v8bf*>(&r) + 1) = *reinterpret_cast<const v8bf*>(row + base + 16);
  return r;
}

// Same A layout but source is fp32 (x), converted to bf16 in registers.
__device__ __forceinline__ v16bf fragA_f32(const float* __restrict__ row,
                                           int kiter, int half) {
  const int base = kiter * 32 + half * 8;
  v16bf r;
#pragma unroll
  for (int i = 0; i < 8; ++i) {
    r[i]     = (__bf16)row[base + i];
    r[i + 8] = (__bf16)row[base + 16 + i];
  }
  return r;
}

// B-matrix fragment (32x16 bf16): lane = N column; lanes 0-15 hold K chunks
// {0-7,16-23}, lanes 16-31 hold {8-15,24-31} of the window (matches A).
__device__ __forceinline__ v16bf fragB_bf16(const __bf16* __restrict__ row,
                                            int kiter, int half) {
  const int base = kiter * 32 + half * 8;
  v16bf r;
  *reinterpret_cast<v8bf*>(&r)       = *reinterpret_cast<const v8bf*>(row + base);
  *(reinterpret_cast<v8bf*>(&r) + 1) = *reinterpret_cast<const v8bf*>(row + base + 16);
  return r;
}

__device__ __forceinline__ float fast_sigmoid(float x) {
  return 1.0f / (1.0f + __expf(-x));
}
__device__ __forceinline__ float fast_tanh(float x) {
  float e = __expf(2.0f * x);
  return (e - 1.0f) / (e + 1.0f);
}

// ---- Tensor Data Mover: 1-D contiguous global -> LDS copy -----------------
// D# per CDNA5 ISA: group0 = {count=1, lds_addr,
// global_addr, type=2}; group1 = {data_size=4B, tensor_dim0 = tile_dim0 =
// tensor_dim0_stride = nbytes/4, workgroup_mask=0}.
__device__ __forceinline__ void tdm_load_1d(unsigned int lds_addr,
                                            const void* gptr,
                                            unsigned int nbytes) {
#if __has_builtin(__builtin_amdgcn_tensor_load_to_lds)
  const unsigned long long ga = (unsigned long long)(size_t)gptr;
  const unsigned int elems = nbytes >> 2;           // 4-byte elements
  v4u g0;
  g0[0] = 1u;                                       // count=1 (valid user D#)
  g0[1] = lds_addr;                                 // LDS byte address
  g0[2] = (unsigned int)ga;                         // global_addr[31:0]
  g0[3] = (unsigned int)((ga >> 32) & 0x1FFFFFFu)   // global_addr[56:32]
          | (2u << 30);                             // type=2 ("image")
  v8i g1 = {};
  g1[0] = (int)(2u << 16);                          // data_size=4B, mask=0
  g1[1] = (int)((elems & 0xFFFFu) << 16);           // tensor_dim0[15:0]
  g1[2] = (int)(elems >> 16);                       // tensor_dim0[31:16]
  g1[3] = (int)((elems & 0xFFFFu) << 16);           // tile_dim0
  g1[5] = (int)elems;                               // tensor_dim0_stride[31:0]
  g1[6] = (int)(elems >> 16);                       // tensor_dim0_stride[47:32]
  v4i gz = {};
#if __clang_major__ >= 23
  v8i gpad = {};
  __builtin_amdgcn_tensor_load_to_lds(g0, g1, gz, gz, gpad, 0);
#else
  __builtin_amdgcn_tensor_load_to_lds(g0, g1, gz, gz, 0);
#endif
#endif
}

// ---------------------------------------------------------------------------
// Init: bf16-convert z/n weight slabs, fold biases, zero h, reset barrier.
// gcol in [0,1024): 0..511 = z gate, 512..1023 = n gate; source row = 512+gcol.
// ---------------------------------------------------------------------------
__global__ void gru_init_kernel(
    const float* __restrict__ wih_f, const float* __restrict__ whh_f,
    const float* __restrict__ bih_f, const float* __restrict__ bhh_f,
    const float* __restrict__ wih_r, const float* __restrict__ whh_r,
    const float* __restrict__ bih_r, const float* __restrict__ bhh_r,
    __bf16* __restrict__ wih_bf, __bf16* __restrict__ whh_bf,
    float* __restrict__ bias_zn, float* __restrict__ hf32,
    __bf16* __restrict__ hbf16, unsigned int* __restrict__ bar) {
  const int tid = blockIdx.x * blockDim.x + threadIdx.x;
  const int nth = gridDim.x * blockDim.x;

  for (int idx = tid; idx < 2 * 1024 * 512; idx += nth) {
    const int dir  = idx >> 19;              // 1024*512 = 2^19
    const int rem  = idx & ((1 << 19) - 1);
    const int gcol = rem >> 9;               // 0..1023
    const int k    = rem & 511;
    const int srow = 512 + gcol;             // skip r rows [0,512)
    const float* wi = dir ? wih_r : wih_f;
    const float* wh = dir ? whh_r : whh_f;
    wih_bf[idx] = (__bf16)wi[srow * 512 + k];
    whh_bf[idx] = (__bf16)wh[srow * 512 + k];
  }
  for (int idx = tid; idx < 2 * 1024; idx += nth) {
    const int dir = idx >> 10;
    const int gcol = idx & 1023;
    const float* bi = dir ? bih_r : bih_f;
    const float* bh = dir ? bhh_r : bhh_f;
    bias_zn[idx] = bi[512 + gcol] + bh[512 + gcol];
  }
  for (int idx = tid; idx < 2 * H_SZ * B_SZ; idx += nth) {
    hf32[idx]  = 0.0f;
    hbf16[idx] = (__bf16)0.0f;
  }
  if (tid == 0) *bar = 0u;
}

// ---------------------------------------------------------------------------
// Phase 1: pre[dir][s][gcol][b] = x[b,s,:] . Wih_zn[dir,gcol,:] + (bih+bhh)
// Grid: (S_SZ, 16). 8 waves/WG, each wave = 16(M) x 64(N) tile, K=512.
// ---------------------------------------------------------------------------
__global__ __launch_bounds__(256) void gru_proj_kernel(
    const float* __restrict__ x, const __bf16* __restrict__ wih_bf,
    const float* __restrict__ bias_zn, float* __restrict__ pre) {
  const int s    = blockIdx.x;
  const int nb   = blockIdx.y;               // 128-column block of N=2048
  const int tid  = threadIdx.x;
  const int wave = tid >> 5, lane = tid & 31;
  const int half = lane >> 4, l16 = lane & 15;
  const int mt   = wave & 3, ng = wave >> 2;
  const int mrow = mt * 16 + l16;            // batch row for A fragment
  const int nbase = nb * 128 + ng * 64;

  const float* arow = x + ((size_t)mrow * S_SZ + s) * I_SZ;
  v8f c0{}, c1{}, c2{}, c3{};
#pragma unroll 4
  for (int k = 0; k < 16; ++k) {
    v16bf a = fragA_f32(arow, k, half);
    v16bf b;
    b = fragB_bf16(wih_bf + (size_t)(nbase +  0 + l16) * 512, k, half); c0 = wmma_bf16(a, b, c0);
    b = fragB_bf16(wih_bf + (size_t)(nbase + 16 + l16) * 512, k, half); c1 = wmma_bf16(a, b, c1);
    b = fragB_bf16(wih_bf + (size_t)(nbase + 32 + l16) * 512, k, half); c2 = wmma_bf16(a, b, c2);
    b = fragB_bf16(wih_bf + (size_t)(nbase + 48 + l16) * 512, k, half); c3 = wmma_bf16(a, b, c3);
  }

  const int mbase = mt * 16 + half * 8;      // C layout: lane holds 8 rows
#pragma unroll
  for (int ns = 0; ns < 4; ++ns) {
    const int n = nbase + ns * 16 + l16;     // global N in [0,2048)
    v8f c = (ns == 0) ? c0 : (ns == 1) ? c1 : (ns == 2) ? c2 : c3;
    const float bv = bias_zn[n];
#pragma unroll
    for (int i = 0; i < 8; ++i) c[i] += bv;
    const int dir = n >> 10, gcol = n & 1023;
    *reinterpret_cast<v8f*>(
        pre + ((((size_t)dir * S_SZ + s) * 1024 + gcol) * B_SZ + mbase)) = c;
  }
}

// ---------------------------------------------------------------------------
// Phase 2: persistent recurrence. 16 WGs: WG w -> dir = w>>3, cols [64w0,64w0+64).
// Whh(z,n) slice staged by TDM and resident in LDS for all 512 steps.
// Waves 0-3: z gate, waves 4-7: n gate; sigmoid(z) exchanged via LDS;
// grid barrier per step.
// ---------------------------------------------------------------------------
__global__ __launch_bounds__(256) void gru_recur_kernel(
    const float* __restrict__ pre, const __bf16* __restrict__ whh_bf,
    float* __restrict__ hf32, __bf16* __restrict__ hbf16,
    float* __restrict__ out, float* __restrict__ hout,
    unsigned int* __restrict__ bar) {
  __shared__ __bf16 lds_wz[64 * 512];        // 64 KB, resident all steps
  __shared__ __bf16 lds_wn[64 * 512];        // 64 KB
  __shared__ float  lds_z[64 * 64];          // 16 KB sigmoid(z) exchange

  const int w    = blockIdx.x;
  const int dir  = w >> 3;
  const int j0   = (w & 7) * 64;             // hidden-column slice
  const int tid  = threadIdx.x;
  const int wave = tid >> 5, lane = tid & 31;
  const int half = lane >> 4, l16 = lane & 15;
  const int isN  = wave >> 2, mt = wave & 3;
  const int mrow = mt * 16 + l16;
  const int mbase = mt * 16 + half * 8;

  // Stage this WG's z and n weight rows (bf16, contiguous 64 KB slabs) to LDS.
  const __bf16* srcz = whh_bf + ((size_t)dir * 1024 + j0) * 512;
  const __bf16* srcn = whh_bf + ((size_t)dir * 1024 + 512 + j0) * 512;
#if __has_builtin(__builtin_amdgcn_tensor_load_to_lds)
  if (wave == 0) {
    tdm_load_1d((unsigned int)(size_t)lds_wz, srcz, 64 * 512 * 2);
    tdm_load_1d((unsigned int)(size_t)lds_wn, srcn, 64 * 512 * 2);
    __builtin_amdgcn_s_wait_tensorcnt(0);
  }
#else
  {
    v8bf* dz = (v8bf*)lds_wz;
    v8bf* dn = (v8bf*)lds_wn;
    for (int c = tid; c < 64 * 512 / 8; c += 256) {
      dz[c] = ((const v8bf*)srcz)[c];
      dn[c] = ((const v8bf*)srcn)[c];
    }
  }
#endif
  __syncthreads();

  const __bf16* wsel = isN ? lds_wn : lds_wz;
  __bf16* hA = hbf16 + (size_t)dir * B_SZ * H_SZ;   // [b][col] bf16 (A matrix)
  float*  hF = hf32  + (size_t)dir * H_SZ * B_SZ;   // [col][b] fp32 (blend)
  const float* preD = pre + (size_t)dir * S_SZ * 1024 * B_SZ;

  for (int step = 0; step < S_SZ; ++step) {
    const int t = dir ? (S_SZ - 1 - step) : step;
    const __bf16* arow = hA + (size_t)mrow * H_SZ;
    v8f c0{}, c1{}, c2{}, c3{};
#pragma unroll 4
    for (int k = 0; k < 16; ++k) {
      v16bf a = fragA_bf16(arow, k, half);
      v16bf b;
      b = fragB_bf16(wsel + (size_t)( 0 + l16) * 512, k, half); c0 = wmma_bf16(a, b, c0);
      b = fragB_bf16(wsel + (size_t)(16 + l16) * 512, k, half); c1 = wmma_bf16(a, b, c1);
      b = fragB_bf16(wsel + (size_t)(32 + l16) * 512, k, half); c2 = wmma_bf16(a, b, c2);
      b = fragB_bf16(wsel + (size_t)(48 + l16) * 512, k, half); c3 = wmma_bf16(a, b, c3);
    }

    const size_t preT = (size_t)t * 1024 * B_SZ;
    if (!isN) {                               // z waves: sigmoid -> LDS
#pragma unroll
      for (int ns = 0; ns < 4; ++ns) {
        const int nl = ns * 16 + l16;
        v8f cc = (ns == 0) ? c0 : (ns == 1) ? c1 : (ns == 2) ? c2 : c3;
        const v8f pz = *reinterpret_cast<const v8f*>(
            preD + preT + (size_t)(j0 + nl) * B_SZ + mbase);
        v8f zv;
#pragma unroll
        for (int i = 0; i < 8; ++i) zv[i] = fast_sigmoid(cc[i] + pz[i]);
        *reinterpret_cast<v8f*>(lds_z + nl * 64 + mbase) = zv;
      }
    }
    __syncthreads();
    if (isN) {                                // n waves: tanh + blend + stores
#pragma unroll
      for (int ns = 0; ns < 4; ++ns) {
        const int nl = ns * 16 + l16;
        const int gj = j0 + nl;
        v8f cc = (ns == 0) ? c0 : (ns == 1) ? c1 : (ns == 2) ? c2 : c3;
        const v8f pn = *reinterpret_cast<const v8f*>(
            preD + preT + (size_t)(512 + gj) * B_SZ + mbase);
        const v8f zv = *reinterpret_cast<const v8f*>(lds_z + nl * 64 + mbase);
        const v8f ho = *reinterpret_cast<const v8f*>(hF + (size_t)gj * B_SZ + mbase);
        v8f hn;
#pragma unroll
        for (int i = 0; i < 8; ++i) {
          float nt = fast_tanh(cc[i] + pn[i]);
          hn[i] = (1.0f - zv[i]) * nt + zv[i] * ho[i];
        }
        *reinterpret_cast<v8f*>(hF + (size_t)gj * B_SZ + mbase) = hn;
#pragma unroll
        for (int i = 0; i < 8; ++i) {
          hA[(size_t)(mbase + i) * H_SZ + gj] = (__bf16)hn[i];
          out[((size_t)(mbase + i) * S_SZ + t) * (2 * H_SZ) + dir * H_SZ + gj] = hn[i];
        }
        if (step == S_SZ - 1) {
#pragma unroll
          for (int i = 0; i < 8; ++i)
            hout[((size_t)dir * B_SZ + (mbase + i)) * H_SZ + gj] = hn[i];
        }
      }
    }
    // Prefetch next step's pre rows into cache while we sync.
    if (step + 1 < S_SZ) {
      const int t2 = dir ? (S_SZ - 2 - step) : (step + 1);
      __builtin_prefetch(preD + ((size_t)t2 * 1024 + (isN ? 512 : 0) + j0 + l16) * B_SZ, 0, 1);
    }

    __threadfence();
    __syncthreads();
    __builtin_amdgcn_s_cluster_barrier();     // NOP when not dispatched as cluster
    if (tid == 0) {
      __hip_atomic_fetch_add(bar, 1u, __ATOMIC_RELEASE, __HIP_MEMORY_SCOPE_AGENT);
      const unsigned int target = (unsigned int)NWG * (unsigned int)(step + 1);
      while (__hip_atomic_load(bar, __ATOMIC_ACQUIRE, __HIP_MEMORY_SCOPE_AGENT) < target) {
        __builtin_amdgcn_s_sleep(2);
      }
    }
    __syncthreads();
  }
}

// ---------------------------------------------------------------------------
extern "C" void kernel_launch(void* const* d_in, const int* in_sizes, int n_in,
                              void* d_out, int out_size, void* d_ws, size_t ws_size,
                              hipStream_t stream) {
  const float* x     = (const float*)d_in[0];
  const float* wih_f = (const float*)d_in[1];
  const float* whh_f = (const float*)d_in[2];
  const float* bih_f = (const float*)d_in[3];
  const float* bhh_f = (const float*)d_in[4];
  const float* wih_r = (const float*)d_in[5];
  const float* whh_r = (const float*)d_in[6];
  const float* bih_r = (const float*)d_in[7];
  const float* bhh_r = (const float*)d_in[8];

  char* ws = (char*)d_ws;
  size_t off = 0;
  float* pre = (float*)(ws + off);        off += (size_t)2 * S_SZ * 1024 * B_SZ * 4; // 256 MB
  __bf16* wih_bf = (__bf16*)(ws + off);   off += (size_t)2 * 1024 * 512 * 2;         // 2 MB
  __bf16* whh_bf = (__bf16*)(ws + off);   off += (size_t)2 * 1024 * 512 * 2;         // 2 MB
  float* bias_zn = (float*)(ws + off);    off += (size_t)2 * 1024 * 4;
  float* hf32 = (float*)(ws + off);       off += (size_t)2 * H_SZ * B_SZ * 4;
  __bf16* hbf16 = (__bf16*)(ws + off);    off += (size_t)2 * B_SZ * H_SZ * 2;
  unsigned int* bar = (unsigned int*)(ws + off); off += 128;

  float* out  = (float*)d_out;
  float* hout = out + (size_t)B_SZ * S_SZ * 2 * H_SZ;

  gru_init_kernel<<<512, 256, 0, stream>>>(wih_f, whh_f, bih_f, bhh_f,
                                           wih_r, whh_r, bih_r, bhh_r,
                                           wih_bf, whh_bf, bias_zn, hf32, hbf16, bar);
  gru_proj_kernel<<<dim3(S_SZ, 16), 256, 0, stream>>>(x, wih_bf, bias_zn, pre);
  gru_recur_kernel<<<NWG, 256, 0, stream>>>(pre, whh_bf, hf32, hbf16, out, hout, bar);
}
